// DeformConv_41704132444209
// MI455X (gfx1250) — matrix operand order repaired and checked by reading, not compiled, not run on hardware
//
#include <hip/hip_runtime.h>
#include <hip/hip_bf16.h>
#include <cstdint>

typedef __attribute__((ext_vector_type(16))) __bf16 v16bf;
typedef __attribute__((ext_vector_type(8)))  float  v8f;

#define HW      64
#define HWPIX   4096
#define CIN     256
#define OCH     256
#define NB      8
#define NTAP    9
#define WT2_ELEMS (9*8*256*32)     /* 589824 bf16 */
#define WT2_BYTES (WT2_ELEMS*2)    /* 1179648 B   */

union ABv { v16bf v; uint4 q[2]; };

__device__ __forceinline__ unsigned short f2bf(float f) {
  unsigned u = __float_as_uint(f);
  u += 0x7FFFu + ((u >> 16) & 1u);          // round-to-nearest-even
  return (unsigned short)(u >> 16);
}

// ---------------------------------------------------------------------------
// Kernel 1: repack conv_w (OC,C,3,3) f32 -> Wt2[tap][cb][oc][cc] bf16
// so A fragments are two aligned b128 global loads per lane.
// ---------------------------------------------------------------------------
__global__ __launch_bounds__(256) void repack_w(const float* __restrict__ conv_w,
                                                unsigned short* __restrict__ wt2) {
  int i = blockIdx.x * 256 + threadIdx.x;
  if (i >= WT2_ELEMS) return;
  int cc  = i & 31;
  int oc  = (i >> 5) & 255;
  int cb  = (i >> 13) & 7;
  int tap = i >> 16;
  int c   = cb * 32 + cc;
  wt2[i] = f2bf(conv_w[oc * 2304 + c * 9 + tap]);
}

// ---------------------------------------------------------------------------
// Kernel 2: fused offset(18ch) + mod(9ch) 3x3 conv, pad=1. One thread per
// pixel of a 16x16 tile, 27 accumulators, LDS-staged halo + weights per c.
// Output layout: offmod[b][27][y][x] f32 (ch 0..17 offset, 18..26 mod).
// ---------------------------------------------------------------------------
__global__ __launch_bounds__(256) void offmod_conv(
    const float* __restrict__ x,
    const float* __restrict__ offset_w, const float* __restrict__ offset_b,
    const float* __restrict__ mod_w,    const float* __restrict__ mod_b,
    float* __restrict__ offmod) {
  __shared__ float xs[18 * 18];
  __shared__ float ws[27 * 9];
  int blk  = blockIdx.x;
  int b    = blk >> 4;
  int tile = blk & 15;
  int ty0 = (tile >> 2) * 16, tx0 = (tile & 3) * 16;
  int tid = threadIdx.x;
  int ly = tid >> 4, lx = tid & 15;
  float acc[27];
#pragma unroll
  for (int o = 0; o < 27; ++o) acc[o] = 0.f;

  for (int c = 0; c < CIN; ++c) {
    for (int i = tid; i < 324; i += 256) {
      int hy = i / 18, hx = i - hy * 18;
      int gy = ty0 + hy - 1, gx = tx0 + hx - 1;
      float v = 0.f;
      if ((unsigned)gy < 64u && (unsigned)gx < 64u)
        v = x[(((size_t)b * CIN + c) << 12) + gy * 64 + gx];
      xs[i] = v;
    }
    for (int i = tid; i < 243; i += 256) {
      int o = i / 9, t = i - o * 9;
      ws[i] = (o < 18) ? offset_w[o * 2304 + c * 9 + t]
                       : mod_w[(o - 18) * 2304 + c * 9 + t];
    }
    __syncthreads();
#pragma unroll
    for (int t = 0; t < 9; ++t) {
      int ky = t / 3, kx = t - ky * 3;
      float xv = xs[(ly + ky) * 18 + lx + kx];
#pragma unroll
      for (int o = 0; o < 27; ++o) acc[o] = fmaf(ws[o * 9 + t], xv, acc[o]);
    }
    __syncthreads();
  }
  int py = ty0 + ly, px = tx0 + lx;
  size_t base = (size_t)b * 27 * HWPIX + py * 64 + px;
#pragma unroll
  for (int o = 0; o < 27; ++o) {
    float bia = (o < 18) ? offset_b[o] : mod_b[o - 18];
    offmod[base + (size_t)o * HWPIX] = acc[o] + bia;
  }
}

// ---------------------------------------------------------------------------
// Kernel 3: fused bilinear-sample + modulation + 256x2304x32768 bf16 GEMM.
// Block = 256 oc x 128 px, 512 threads = 16 waves (wave32), arranged 4(M)x4(N),
// each wave owns 4x2 wmma 16x16 tiles (64 f32 accum VGPRs/lane).
// Sampling is done ONCE per (px, c) since the full M=256 is covered per block.
// ---------------------------------------------------------------------------
__global__ __launch_bounds__(512) void deform_gemm(
    const float* __restrict__ x,
    const unsigned short* __restrict__ wt2,
    const float* __restrict__ offmod,
    const float* __restrict__ conv_b,
    float* __restrict__ out) {
  __shared__ int   s_off[NTAP * 128 * 4];
  __shared__ float s_wt [NTAP * 128 * 4];
  __shared__ __align__(16) __bf16 Bs[128 * 40];  // row stride 40 elems -> 80B

  int tid = threadIdx.x;
  int pt = blockIdx.x & 31;
  int b  = blockIdx.x >> 5;
  int pxBase = pt * 128;

  // --- per-(tap, pixel) bilinear corner offsets and mod-folded weights -----
  for (int t = tid; t < NTAP * 128; t += 512) {
    int px  = t & 127;
    int tap = t >> 7;
    int p = pxBase + px;
    int y = p >> 6, xx = p & 63;
    int ky = tap / 3, kx = tap - ky * 3;
    int py = y + ky - 1, qx = xx + kx - 1;
    int   o00 = 0, o01 = 0, o10 = 0, o11 = 0;
    float w00 = 0.f, w01 = 0.f, w10 = 0.f, w11 = 0.f;
    if ((unsigned)py < 64u && (unsigned)qx < 64u) {
      int spx = py * 64 + qx;
      size_t ob = (size_t)b * 27 * HWPIX + spx;
      float gx = offmod[ob + (size_t)tap * HWPIX]        + (qx * (2.0f / 63.0f) - 1.0f);
      float gy = offmod[ob + (size_t)(9 + tap) * HWPIX]  + (py * (2.0f / 63.0f) - 1.0f);
      float m  = offmod[ob + (size_t)(18 + tap) * HWPIX];
      gx = fminf(fmaxf(gx, -1.f), 1.f);
      gy = fminf(fmaxf(gy, -1.f), 1.f);
      float ix = (gx + 1.f) * 0.5f * 63.f;
      float iy = (gy + 1.f) * 0.5f * 63.f;
      float x0 = floorf(ix), y0 = floorf(iy);
      float wx = ix - x0, wy = iy - y0;
      int x0i = min(max((int)x0, 0), 63);
      int x1i = min(x0i + 1, 63);
      int y0i = min(max((int)y0, 0), 63);
      int y1i = min(y0i + 1, 63);
      o00 = y0i * 64 + x0i; o01 = y0i * 64 + x1i;
      o10 = y1i * 64 + x0i; o11 = y1i * 64 + x1i;
      w00 = (1.f - wx) * (1.f - wy) * m; w01 = wx * (1.f - wy) * m;
      w10 = (1.f - wx) * wy * m;         w11 = wx * wy * m;
    }
    int i4 = t * 4;
    s_off[i4] = o00; s_off[i4 + 1] = o01; s_off[i4 + 2] = o10; s_off[i4 + 3] = o11;
    s_wt [i4] = w00; s_wt [i4 + 1] = w01; s_wt [i4 + 2] = w10; s_wt [i4 + 3] = w11;
  }
  __syncthreads();

  int lane  = tid & 31;
  int wv    = tid >> 5;          // 0..15
  int m0w   = (wv >> 2) * 64;    // 0 / 64 / 128 / 192
  int n0w   = (wv & 3) * 32;     // 0..96
  int mlane = lane & 15;
  int hi    = lane >> 4;         // 0/1 half-wave

  v8f acc[4][2];
#pragma unroll
  for (int mt = 0; mt < 4; ++mt)
#pragma unroll
    for (int nt = 0; nt < 2; ++nt)
#pragma unroll
      for (int i = 0; i < 8; ++i) acc[mt][nt][i] = 0.f;

  int pxS     = tid >> 2;        // 0..127 : pixel this thread stages
  int quarter = tid & 3;         // 8 channels per thread

  for (int it = 0; it < 72; ++it) {          // it = tap*8 + cb, K-chunk of 32
    int tap = it >> 3;
    int cb  = it & 7;
    // ---- stage B tile: sampled*mod values, bf16, Bs[px][cc] ---------------
    {
      int i4 = ((tap << 7) + pxS) * 4;
      int o00 = s_off[i4], o01 = s_off[i4 + 1], o10 = s_off[i4 + 2], o11 = s_off[i4 + 3];
      float w00 = s_wt[i4], w01 = s_wt[i4 + 1], w10 = s_wt[i4 + 2], w11 = s_wt[i4 + 3];
      const float* xp = x + (((size_t)(b * CIN + cb * 32 + quarter * 8)) << 12);
      __bf16* brow = &Bs[pxS * 40 + quarter * 8];
#pragma unroll
      for (int j = 0; j < 8; ++j) {
        const float* pc = xp + ((size_t)j << 12);
        float v = w00 * pc[o00] + w01 * pc[o01] + w10 * pc[o10] + w11 * pc[o11];
        brow[j] = (__bf16)v;
      }
    }
    if (it + 1 < 72) {  // prefetch next 16KB weight slab (uniform branch)
      const unsigned short* ns = wt2 + (((size_t)(it + 1)) << 13);
      __builtin_prefetch((const void*)(ns + (size_t)tid * 16), 0, 1);
    }
    __syncthreads();

    // ---- fragments + WMMA -------------------------------------------------
    ABv Bf[2];
#pragma unroll
    for (int nt = 0; nt < 2; ++nt) {
      const __bf16* bp = &Bs[(n0w + nt * 16 + mlane) * 40 + hi * 8];
      Bf[nt].q[0] = *(const uint4*)bp;
      Bf[nt].q[1] = *(const uint4*)(bp + 16);
    }
    const unsigned short* aSlab = wt2 + (((size_t)it) << 13);
#pragma unroll
    for (int mt = 0; mt < 4; ++mt) {
      int m = m0w + mt * 16 + mlane;
      ABv Af;
      Af.q[0] = *(const uint4*)(aSlab + m * 32 + hi * 8);        // K 0..7  / 8..15
      Af.q[1] = *(const uint4*)(aSlab + m * 32 + 16 + hi * 8);   // K 16..23 / 24..31
      acc[mt][0] = __builtin_amdgcn_wmma_f32_16x16x32_bf16(
          false, Af.v, false, Bf[0].v, (short)0, acc[mt][0], false, false);
      acc[mt][1] = __builtin_amdgcn_wmma_f32_16x16x32_bf16(
          false, Af.v, false, Bf[1].v, (short)0, acc[mt][1], false, false);
    }
    __syncthreads();
  }

  // ---- epilogue: bias + store (C/D layout: VGPR i -> M = hi*8 + i) --------
#pragma unroll
  for (int mt = 0; mt < 4; ++mt) {
#pragma unroll
    for (int nt = 0; nt < 2; ++nt) {
      int col = pxBase + n0w + nt * 16 + mlane;
      int rowBase = m0w + mt * 16 + hi * 8;
#pragma unroll
      for (int i = 0; i < 8; ++i) {
        int row = rowBase + i;
        out[(((size_t)(b * OCH + row)) << 12) + col] = acc[mt][nt][i] + conv_b[row];
      }
    }
  }
}

// ---------------------------------------------------------------------------
extern "C" void kernel_launch(void* const* d_in, const int* in_sizes, int n_in,
                              void* d_out, int out_size, void* d_ws, size_t ws_size,
                              hipStream_t stream) {
  (void)in_sizes; (void)n_in; (void)out_size; (void)ws_size;
  const float* x        = (const float*)d_in[0];
  const float* conv_w   = (const float*)d_in[1];
  const float* conv_b   = (const float*)d_in[2];
  const float* offset_w = (const float*)d_in[3];
  const float* offset_b = (const float*)d_in[4];
  const float* mod_w    = (const float*)d_in[5];
  const float* mod_b    = (const float*)d_in[6];
  float* out = (float*)d_out;

  unsigned short* wt2 = (unsigned short*)d_ws;                    // 1.13 MB
  float* offmod = (float*)((char*)d_ws + WT2_BYTES);              // 3.38 MB

  repack_w<<<(WT2_ELEMS + 255) / 256, 256, 0, stream>>>(conv_w, wt2);
  offmod_conv<<<NB * 16, 256, 0, stream>>>(x, offset_w, offset_b, mod_w, mod_b, offmod);
  deform_gemm<<<NB * 32, 512, 0, stream>>>(x, wt2, offmod, conv_b, out);
}
